// GraphNet_48404281426504
// MI455X (gfx1250) — matrix operand order repaired and checked by reading, not compiled, not run on hardware
//
#include <hip/hip_runtime.h>

// ---------------------------------------------------------------------------
// MPNN (NNConv + GRU + Set2Set) for MI455X / gfx1250, wave32, bf16 WMMA.
//  * All GEMM operands pre-converted to bf16, K padded to mult-of-32, B stored
//    transposed (NxKp) -> fragment loads are branch-free b128 vector loads.
//  * Row/col clamping instead of predication (garbage rows/cols never stored).
//  * One 16x32 C tile per wave: 1 A frag + 2 B frags -> 2 WMMA per k-step.
//  * fnet edge tensor (50000x64x64) computed once, stored bf16 (410 MB).
//  * Per-edge matvec + scatter-add stays memory-bound VALU + f32 atomics.
// ---------------------------------------------------------------------------

typedef __attribute__((ext_vector_type(16))) __bf16 bf16x16;
typedef __attribute__((ext_vector_type(8)))  __bf16 bf16x8;
typedef __attribute__((ext_vector_type(2)))  __bf16 bf16x2;
typedef __attribute__((ext_vector_type(8)))  float  f32x8;

#define N_NODES 25000
#define N_EDGES 50000
#define D       64
#define B_GRAPHS 800

__device__ inline float sigm(float x) { return 1.0f / (1.0f + __expf(-x)); }

// order-preserving float <-> uint for atomicMax-based segment max
__device__ inline unsigned fenc(float f) {
    unsigned u = __float_as_uint(f);
    return (u & 0x80000000u) ? ~u : (u | 0x80000000u);
}
__device__ inline float fdec(unsigned u) {
    unsigned v = (u & 0x80000000u) ? (u & 0x7FFFFFFFu) : ~u;
    return __uint_as_float(v);
}

__device__ inline f32x8 wmma_bf16(bf16x16 a, bf16x16 b, f32x8 c) {
    return __builtin_amdgcn_wmma_f32_16x16x32_bf16(
        false, a, false, b, (short)0, c, false, false);
}

// ---------------- branch-free bf16 fragment loaders (ISA 7.12.2) -----------
// A (MxKp row-major bf16): lane l holds row l&15; e<8 -> K=k0+(l>>4)*8+e,
// e>=8 -> +16. Two contiguous 16B loads.
__device__ inline bf16x16 load_a(const __bf16* __restrict__ A, int lda, int M,
                                 int row0, int k0, int lane) {
    int m = row0 + (lane & 15);
    m = m < M ? m : M - 1;                       // clamp: garbage row never stored
    const __bf16* r = A + (long)m * lda + k0 + (lane >> 4) * 8;
    bf16x8 lo = *(const bf16x8*)r;
    bf16x8 hi = *(const bf16x8*)(r + 16);
    return __builtin_shufflevector(lo, hi, 0, 1, 2, 3, 4, 5, 6, 7,
                                   8, 9, 10, 11, 12, 13, 14, 15);
}

// B stored transposed (NxKp row-major bf16): lane l holds col l&15;
// e<8 -> K=k0+(l>>4)*8+e, e>=8 -> +16. Two contiguous 16B loads.
__device__ inline bf16x16 load_b(const __bf16* __restrict__ Bt, int ldb, int N,
                                 int col0, int k0, int lane) {
    int n = col0 + (lane & 15);
    n = n < N ? n : N - 1;                       // clamp: garbage col never stored
    const __bf16* r = Bt + (long)n * ldb + k0 + (lane >> 4) * 8;
    bf16x8 lo = *(const bf16x8*)r;
    bf16x8 hi = *(const bf16x8*)(r + 16);
    return __builtin_shufflevector(lo, hi, 0, 1, 2, 3, 4, 5, 6, 7,
                                   8, 9, 10, 11, 12, 13, 14, 15);
}

template <int ACT, bool OBF>
__device__ inline void store_tile(const f32x8& acc, void* __restrict__ C,
                                  const float* __restrict__ bias,
                                  int M, int N, int ldc, int row0, int col0, int lane) {
    int n = col0 + (lane & 15);
    if (n >= N) return;
    float bv = bias ? bias[n] : 0.0f;
    int half = lane >> 4;
#pragma unroll
    for (int r = 0; r < 8; ++r) {
        int m = row0 + r + 8 * half;
        if (m < M) {
            float v = acc[r] + bv;
            if (ACT) v = fmaxf(v, 0.0f);
            if (OBF) ((__bf16*)C)[(long)m * ldc + n] = (__bf16)v;
            else     ((float*)C)[(long)m * ldc + n]  = v;
        }
    }
}

// ---------------- WMMA GEMM: C = act(A @ Bt^T + bias) ----------------------
// A: MxKp bf16 (zero-padded), Bt: NxKp bf16 (transposed, zero-padded),
// Kp multiple of 32, N multiple of 32. One 16x32 tile per wave.
template <int ACT, bool OBF>
__global__ __launch_bounds__(256) void gemm_bf16(const __bf16* __restrict__ A,
                                                 const __bf16* __restrict__ Bt,
                                                 const float* __restrict__ bias,
                                                 void* __restrict__ C,
                                                 int M, int N, int Kp, int ldc) {
    int wave = (int)((blockIdx.x * blockDim.x + threadIdx.x) >> 5);
    int lane = threadIdx.x & 31;
    int tilesN = N >> 5;                 // N is a multiple of 32
    int tilesM = (M + 15) >> 4;
    int tm = wave / tilesN;
    int tn = wave - tm * tilesN;
    if (tm >= tilesM) return;            // whole-wave exit: EXEC all-1s at WMMA
    int row0 = tm * 16, col0 = tn * 32;

    f32x8 acc0 = {}, acc1 = {};
    for (int k0 = 0; k0 < Kp; k0 += 32) {
        bf16x16 a  = load_a(A, Kp, M, row0, k0, lane);
        bf16x16 b0 = load_b(Bt, Kp, N, col0, k0, lane);
        bf16x16 b1 = load_b(Bt, Kp, N, col0 + 16, k0, lane);
        acc0 = wmma_bf16(a, b0, acc0);
        acc1 = wmma_bf16(a, b1, acc1);
    }
    store_tile<ACT, OBF>(acc0, C, bias, M, N, ldc, row0, col0, lane);
    store_tile<ACT, OBF>(acc1, C, bias, M, N, ldc, row0, col0 + 16, lane);
}

template <int ACT, bool OBF>
static void launch_gemm(const __bf16* A, const __bf16* Bt, const float* bias,
                        void* C, int M, int N, int Kp, int ldc, hipStream_t s) {
    long tiles = (long)((M + 15) / 16) * (N / 32);
    int blocks = (int)((tiles + 7) / 8);  // 8 waves / 256-thread block
    gemm_bf16<ACT, OBF><<<blocks, 256, 0, s>>>(A, Bt, bias, C, M, N, Kp, ldc);
}

// ---------------- operand conversion kernels -------------------------------
// fp32 MxK -> bf16 MxKp (zero pad)
__global__ void cvt_pad(const float* __restrict__ A, __bf16* __restrict__ O,
                        int K, int Kp) {
    int k = blockIdx.x * blockDim.x + threadIdx.x;
    int m = blockIdx.y;
    if (k >= Kp) return;
    O[(long)m * Kp + k] = (k < K) ? (__bf16)A[(long)m * K + k] : (__bf16)0.0f;
}
// fp32 KxN -> bf16 NxKp (transpose + zero pad)
__global__ void cvt_t_pad(const float* __restrict__ B, __bf16* __restrict__ O,
                          int K, int N, int Kp) {
    int k = blockIdx.x * blockDim.x + threadIdx.x;
    int n = blockIdx.y;
    if (k >= Kp) return;
    O[(long)n * Kp + k] = (k < K) ? (__bf16)B[(long)k * N + n] : (__bf16)0.0f;
}

// ---------------- per-edge matvec + scatter-add (memory bound) -------------
__global__ __launch_bounds__(256) void edge_message(const float* __restrict__ h,
                                                    const __bf16* __restrict__ ew,
                                                    const int* __restrict__ src,
                                                    const int* __restrict__ dst,
                                                    float* __restrict__ agg, int E) {
    int e = (int)((blockIdx.x * blockDim.x + threadIdx.x) >> 5);
    int lane = threadIdx.x & 31;
    if (e >= E) return;
    int s = src[e];
    int d = dst[e];
    const float* hs = h + (long)s * D;
    const bf16x2* W = (const bf16x2*)(ew + (long)e * (D * D));
    float a0 = 0.0f, a1 = 0.0f;
#pragma unroll 8
    for (int k = 0; k < D; ++k) {
        float hv = hs[k];                 // wave-uniform, L0 broadcast
        bf16x2 w = W[k * 32 + lane];      // coalesced 128B per row
        a0 += hv * (float)w[0];
        a1 += hv * (float)w[1];
    }
    int c = lane * 2;
    atomicAdd(&agg[(long)d * D + c], a0);
    atomicAdd(&agg[(long)d * D + c + 1], a1);
}

// m = relu(agg + h@Wroot + bconv) -> bf16 (GEMM A operand)
__global__ void nnconv_combine(const float* __restrict__ agg,
                               const float* __restrict__ hroot,
                               const float* __restrict__ bias,
                               __bf16* __restrict__ out, long total) {
    long idx = blockIdx.x * (long)blockDim.x + threadIdx.x;
    if (idx >= total) return;
    out[idx] = (__bf16)fmaxf(agg[idx] + hroot[idx] + bias[idx & (D - 1)], 0.0f);
}

// GRU elementwise combine; updates fp32 h in place + bf16 shadow
__global__ void gru_combine(const float* __restrict__ gi,
                            const float* __restrict__ gh,
                            float* __restrict__ h, __bf16* __restrict__ h_bf,
                            long total) {
    long idx = blockIdx.x * (long)blockDim.x + threadIdx.x;
    if (idx >= total) return;
    long row = idx >> 6;
    int c = idx & (D - 1);
    const float* gir = gi + row * (3 * D);
    const float* ghr = gh + row * (3 * D);
    float r = sigm(gir[c] + ghr[c]);
    float z = sigm(gir[D + c] + ghr[D + c]);
    float n = tanhf(gir[2 * D + c] + r * ghr[2 * D + c]);
    float hn = (1.0f - z) * n + z * h[idx];
    h[idx] = hn;
    h_bf[idx] = (__bf16)hn;
}

// ---------------- Set2Set ---------------------------------------------------

__global__ __launch_bounds__(256) void lstm_step(float* __restrict__ qstar,
                                                 float* __restrict__ hl,
                                                 float* __restrict__ cl,
                                                 const float* __restrict__ Wih,
                                                 const float* __restrict__ Whh,
                                                 const float* __restrict__ bih,
                                                 const float* __restrict__ bhh) {
    __shared__ float sx[2 * D];
    __shared__ float sh[D];
    __shared__ float sg[4 * D];
    int b = blockIdx.x;
    int t = threadIdx.x;  // 256
    if (t < 2 * D) sx[t] = qstar[(long)b * 2 * D + t];
    if (t >= 2 * D && t < 3 * D) sh[t - 2 * D] = hl[(long)b * D + (t - 2 * D)];
    __syncthreads();
    float g = bih[t] + bhh[t];
    const float* wi = Wih + (long)t * (2 * D);
    const float* wh = Whh + (long)t * D;
#pragma unroll 4
    for (int k = 0; k < 2 * D; ++k) g += sx[k] * wi[k];
#pragma unroll 4
    for (int k = 0; k < D; ++k) g += sh[k] * wh[k];
    sg[t] = g;
    __syncthreads();
    if (t < D) {
        float i = sigm(sg[t]);
        float f = sigm(sg[D + t]);
        float gg = tanhf(sg[2 * D + t]);
        float o = sigm(sg[3 * D + t]);
        float c2 = f * cl[(long)b * D + t] + i * gg;
        float h2 = o * tanhf(c2);
        cl[(long)b * D + t] = c2;
        hl[(long)b * D + t] = h2;
        qstar[(long)b * 2 * D + t] = h2;
    }
}

__global__ void seg_init(unsigned* __restrict__ mkey, float* __restrict__ denom,
                         float* __restrict__ qstar, int B) {
    int b = blockIdx.x * blockDim.x + threadIdx.x;
    if (b >= B) return;
    mkey[b] = fenc(-3.4e38f);
    denom[b] = 0.0f;
#pragma unroll 4
    for (int c = 0; c < D; ++c) qstar[(long)b * 2 * D + D + c] = 0.0f;
}

__global__ __launch_bounds__(256) void attn_score(const float* __restrict__ h,
                                                  const float* __restrict__ q,
                                                  const int* __restrict__ batch,
                                                  float* __restrict__ e,
                                                  unsigned* __restrict__ mkey, int Nn) {
    int n = (int)((blockIdx.x * blockDim.x + threadIdx.x) >> 5);
    int lane = threadIdx.x & 31;
    if (n >= Nn) return;
    int b = batch[n];
    const float* hr = h + (long)n * D;
    const float* qr = q + (long)b * D;
    float p = hr[2 * lane] * qr[2 * lane] + hr[2 * lane + 1] * qr[2 * lane + 1];
#pragma unroll
    for (int off = 16; off > 0; off >>= 1) p += __shfl_down(p, off, 32);
    if (lane == 0) {
        e[n] = p;
        atomicMax(&mkey[b], fenc(p));
    }
}

__global__ void attn_exp(const float* __restrict__ e, const unsigned* __restrict__ mkey,
                         const int* __restrict__ batch, float* __restrict__ a,
                         float* __restrict__ denom, int Nn) {
    int n = blockIdx.x * blockDim.x + threadIdx.x;
    if (n >= Nn) return;
    int b = batch[n];
    float v = __expf(e[n] - fdec(mkey[b]));
    a[n] = v;
    atomicAdd(&denom[b], v);
}

__global__ void attn_accum(const float* __restrict__ h, const float* __restrict__ a,
                           const float* __restrict__ denom, const int* __restrict__ batch,
                           float* __restrict__ qstar, int Nn) {
    long idx = blockIdx.x * (long)blockDim.x + threadIdx.x;
    if (idx >= (long)Nn * D) return;
    int n = (int)(idx >> 6);
    int c = (int)(idx & (D - 1));
    int b = batch[n];
    float coef = a[n] / denom[b];
    atomicAdd(&qstar[(long)b * 2 * D + D + c], coef * h[idx]);
}

__global__ void head_final(const float* __restrict__ tmp, const float* __restrict__ Wf2,
                           const float* __restrict__ bf2, float* __restrict__ out, int B) {
    int b = blockIdx.x * blockDim.x + threadIdx.x;
    if (b >= B) return;
    float s = bf2[0];
    const float* tr = tmp + (long)b * D;
#pragma unroll 4
    for (int k = 0; k < D; ++k) s += tr[k] * Wf2[k];
    out[b] = s;
}

// ---------------------------------------------------------------------------

static inline size_t align256(size_t x) { return (x + 255) & ~(size_t)255; }

extern "C" void kernel_launch(void* const* d_in, const int* in_sizes, int n_in,
                              void* d_out, int out_size, void* d_ws, size_t ws_size,
                              hipStream_t stream) {
    (void)in_sizes; (void)n_in; (void)out_size; (void)ws_size;
    const float* x         = (const float*)d_in[0];
    const float* edge_attr = (const float*)d_in[1];
    const int*   edge_index= (const int*)d_in[2];
    const int*   batch     = (const int*)d_in[3];
    const float* Wp   = (const float*)d_in[4];
    const float* bp   = (const float*)d_in[5];
    const float* W1   = (const float*)d_in[6];
    const float* b1   = (const float*)d_in[7];
    const float* W2   = (const float*)d_in[8];
    const float* b2   = (const float*)d_in[9];
    const float* Wroot= (const float*)d_in[10];
    const float* bconv= (const float*)d_in[11];
    const float* Wih  = (const float*)d_in[12];
    const float* Whh  = (const float*)d_in[13];
    const float* bih  = (const float*)d_in[14];
    const float* bhh  = (const float*)d_in[15];
    const float* Wih_l= (const float*)d_in[16];
    const float* Whh_l= (const float*)d_in[17];
    const float* bih_l= (const float*)d_in[18];
    const float* bhh_l= (const float*)d_in[19];
    const float* Wf1  = (const float*)d_in[20];
    const float* bf1  = (const float*)d_in[21];
    const float* Wf2  = (const float*)d_in[22];
    const float* bf2  = (const float*)d_in[23];
    float* out = (float*)d_out;

    const int* src = edge_index;
    const int* dst = edge_index + N_EDGES;

    char* p = (char*)d_ws;
    size_t off = 0;
    auto alloc = [&](size_t bytes) { void* q = p + off; off += align256(bytes); return q; };
    // fp32 state
    float*  h     = (float*)alloc((size_t)N_NODES * D * 4);
    float*  agg   = (float*)alloc((size_t)N_NODES * D * 4);
    float*  hroot = (float*)alloc((size_t)N_NODES * D * 4);
    float*  gi    = (float*)alloc((size_t)N_NODES * 3 * D * 4);
    float*  gh    = (float*)alloc((size_t)N_NODES * 3 * D * 4);
    float*  qstar = (float*)alloc((size_t)B_GRAPHS * 2 * D * 4);
    float*  hl    = (float*)alloc((size_t)B_GRAPHS * D * 4);
    float*  cl    = (float*)alloc((size_t)B_GRAPHS * D * 4);
    float*  escore= (float*)alloc((size_t)N_NODES * 4);
    float*  ascore= (float*)alloc((size_t)N_NODES * 4);
    unsigned* mkey= (unsigned*)alloc((size_t)B_GRAPHS * 4);
    float*  denom = (float*)alloc((size_t)B_GRAPHS * 4);
    float*  thead = (float*)alloc((size_t)B_GRAPHS * D * 4);
    // bf16 GEMM operands (K padded to mult of 32)
    __bf16* x_bf   = (__bf16*)alloc((size_t)N_NODES * 96 * 2);    // 74 -> 96
    __bf16* ea_bf  = (__bf16*)alloc((size_t)N_EDGES * 32 * 2);    // 12 -> 32
    __bf16* t_bf   = (__bf16*)alloc((size_t)N_EDGES * 128 * 2);
    __bf16* h_bf   = (__bf16*)alloc((size_t)N_NODES * D * 2);
    __bf16* m_bf   = (__bf16*)alloc((size_t)N_NODES * D * 2);
    __bf16* qs_bf  = (__bf16*)alloc((size_t)B_GRAPHS * 2 * D * 2);
    __bf16* ew     = (__bf16*)alloc((size_t)N_EDGES * D * D * 2); // 410 MB
    // transposed/padded bf16 weights
    __bf16* WpT    = (__bf16*)alloc((size_t)64 * 96 * 2);
    __bf16* W1T    = (__bf16*)alloc((size_t)128 * 32 * 2);
    __bf16* W2T    = (__bf16*)alloc((size_t)4096 * 128 * 2);
    __bf16* WrootT = (__bf16*)alloc((size_t)64 * 64 * 2);
    __bf16* WihB   = (__bf16*)alloc((size_t)192 * 64 * 2);
    __bf16* WhhB   = (__bf16*)alloc((size_t)192 * 64 * 2);
    __bf16* Wf1T   = (__bf16*)alloc((size_t)64 * 128 * 2);

    const long ND = (long)N_NODES * D;

    // --- operand conversion (tiny) ---
    cvt_pad  <<<dim3(1, N_NODES), 96,  0, stream>>>(x, x_bf, 74, 96);
    cvt_pad  <<<dim3(1, N_EDGES), 32,  0, stream>>>(edge_attr, ea_bf, 12, 32);
    cvt_t_pad<<<dim3(1, 64),      96,  0, stream>>>(Wp, WpT, 74, 64, 96);
    cvt_t_pad<<<dim3(1, 128),     32,  0, stream>>>(W1, W1T, 12, 128, 32);
    cvt_t_pad<<<dim3(1, 4096),    128, 0, stream>>>(W2, W2T, 128, 4096, 128);
    cvt_t_pad<<<dim3(1, 64),      64,  0, stream>>>(Wroot, WrootT, 64, 64, 64);
    cvt_pad  <<<dim3(1, 192),     64,  0, stream>>>(Wih, WihB, 64, 64);   // already NxK
    cvt_pad  <<<dim3(1, 192),     64,  0, stream>>>(Whh, WhhB, 64, 64);   // already NxK
    cvt_t_pad<<<dim3(1, 64),      128, 0, stream>>>(Wf1, Wf1T, 128, 64, 128);

    // --- node projection: h = relu(x @ Wp + bp) (WMMA) ---
    launch_gemm<1, false>(x_bf, WpT, bp, h, N_NODES, D, 96, D, stream);
    cvt_pad<<<dim3(1, N_NODES), 64, 0, stream>>>(h, h_bf, 64, 64);

    // --- edge network (once): t = relu(ea@W1+b1); ew = t@W2+b2 (bf16) ---
    launch_gemm<1, true >(ea_bf, W1T, b1, t_bf, N_EDGES, 128, 32, 128, stream);
    launch_gemm<0, true >(t_bf, W2T, b2, ew, N_EDGES, D * D, 128, D * D, stream);

    // --- 3 message-passing layers ---
    for (int layer = 0; layer < 3; ++layer) {
        hipMemsetAsync(agg, 0, (size_t)ND * 4, stream);
        edge_message<<<(N_EDGES + 7) / 8, 256, 0, stream>>>(h, ew, src, dst, agg, N_EDGES);
        launch_gemm<0, false>(h_bf, WrootT, nullptr, hroot, N_NODES, D, D, D, stream);
        nnconv_combine<<<(int)((ND + 255) / 256), 256, 0, stream>>>(agg, hroot, bconv, m_bf, ND);
        launch_gemm<0, false>(m_bf, WihB, bih, gi, N_NODES, 3 * D, D, 3 * D, stream);
        launch_gemm<0, false>(h_bf, WhhB, bhh, gh, N_NODES, 3 * D, D, 3 * D, stream);
        gru_combine<<<(int)((ND + 255) / 256), 256, 0, stream>>>(gi, gh, h, h_bf, ND);
    }

    // --- Set2Set readout ---
    hipMemsetAsync(qstar, 0, (size_t)B_GRAPHS * 2 * D * 4, stream);
    hipMemsetAsync(hl, 0, (size_t)B_GRAPHS * D * 4, stream);
    hipMemsetAsync(cl, 0, (size_t)B_GRAPHS * D * 4, stream);
    for (int step = 0; step < 6; ++step) {
        lstm_step<<<B_GRAPHS, 256, 0, stream>>>(qstar, hl, cl, Wih_l, Whh_l, bih_l, bhh_l);
        seg_init<<<(B_GRAPHS + 255) / 256, 256, 0, stream>>>(mkey, denom, qstar, B_GRAPHS);
        attn_score<<<(N_NODES + 7) / 8, 256, 0, stream>>>(h, hl, batch, escore, mkey, N_NODES);
        attn_exp<<<(N_NODES + 255) / 256, 256, 0, stream>>>(escore, mkey, batch, ascore, denom, N_NODES);
        attn_accum<<<(int)((ND + 255) / 256), 256, 0, stream>>>(h, ascore, denom, batch, qstar, N_NODES);
    }

    // --- head: out = relu(qstar @ Wf1 + bf1) @ Wf2 + bf2 ---
    cvt_pad<<<dim3(1, B_GRAPHS), 128, 0, stream>>>(qstar, qs_bf, 128, 128);
    launch_gemm<1, false>(qs_bf, Wf1T, bf1, thead, B_GRAPHS, D, 128, D, stream);
    head_final<<<(B_GRAPHS + 255) / 256, 256, 0, stream>>>(thead, Wf2, bf2, out, B_GRAPHS);
}
